// NotearsMLP_18184891531694
// MI455X (gfx1250) — matrix operand order repaired and checked by reading, not compiled, not run on hardware
//
#include <hip/hip_runtime.h>

// ---------------------------------------------------------------------------
// NotearsMLP fused forward for MI455X (gfx1250, wave32).
//
// out[n,k] = maybe_sigmoid( sum_h sigmoid( sum_d x[n,d]*W0[k,h,d] + b0[k,h] )
//                           * W1[k,0,h] + b1[k,0] )
//
// Compute-bound: 34.4 GFLOP vs ~16 MB traffic (AI ~2150 FLOP/B).
// fp32-accurate GEMM via f16 hi/lo split on the wide matrix pipe:
//   a*b ~= a_hi*b_hi + a_hi*b_lo + a_lo*b_hi   (error ~2^-22, f32 accum)
// 12x V_WMMA_F32_16X16X32_F16 per 16x16x128 tile, two accumulator chains.
// ROWS=128 rows of x per block amortizes the per-block W0 load+split
// prologue (halves W0 L2 traffic to ~0.5 GB and prologue executions).
// Epilogue reduction uses DPP row_shr adds (VALU, co-executes with WMMA).
// ---------------------------------------------------------------------------

typedef _Float16 v4h  __attribute__((ext_vector_type(4)));
typedef _Float16 v8h  __attribute__((ext_vector_type(8)));
typedef _Float16 v16h __attribute__((ext_vector_type(16)));
typedef float    v8f  __attribute__((ext_vector_type(8)));

#define N_SAMP 8192
#define D_TOW  128            // towers (k) == input dim (d)
#define D_HID  128            // hidden width (h)
#define ROWS   128            // sample rows per block
#define TILES  (ROWS / 16)
#define KSTEPS (D_TOW / 32)   // K=128 in steps of 32

__device__ __forceinline__ void f32_split(float f, _Float16& hi, _Float16& lo) {
    hi = (_Float16)f;                 // RNE: top ~11 mantissa bits
    lo = (_Float16)(f - (float)hi);   // residual: next ~11 bits
}

__device__ __forceinline__ float fast_sigmoid(float v) {
    // v_mul + v_exp_f32 + v_add + v_rcp_f32  (no IEEE divide sequence)
    return __builtin_amdgcn_rcpf(1.0f + __expf(-v));
}

// v_add_f32 with DPP row_shr:N source (zero shifted-in lanes).  After
// N=1,2,4,8 the last lane of each 16-lane row group holds the group sum.
template <int CTRL>
__device__ __forceinline__ float add_dpp_shr(float v) {
    int s = __builtin_amdgcn_update_dpp(0, __float_as_int(v), CTRL,
                                        0xf, 0xf, true);
    return v + __int_as_float(s);
}

__device__ __forceinline__ float row_reduce16(float v) {
    v = add_dpp_shr<0x111>(v);   // row_shr:1
    v = add_dpp_shr<0x112>(v);   // row_shr:2
    v = add_dpp_shr<0x114>(v);   // row_shr:4
    v = add_dpp_shr<0x118>(v);   // row_shr:8
    return v;                    // valid in lanes 15 and 31
}

// --- binary-input flag ------------------------------------------------------
__global__ void nm_init_flag(int* flag) { *flag = 1; }

__global__ void nm_binary_check(const float* __restrict__ x, int n, int* flag) {
    int i = blockIdx.x * blockDim.x + threadIdx.x;
    bool bad = false;
    if (i < n) {
        float v = x[i];
        bad = !(v == 0.0f || v == 1.0f);
    }
    if (__any(bad) && ((threadIdx.x & 31) == 0)) atomicAnd(flag, 0);
}

// --- fused 2-layer tower kernel --------------------------------------------
__global__ __launch_bounds__(256)
void nm_fused_kernel(const float* __restrict__ x,   // [N, D]
                     const float* __restrict__ W0,  // [D, D1, D]
                     const float* __restrict__ b0,  // [D, D1]
                     const float* __restrict__ W1,  // [D, 1, D1]
                     const float* __restrict__ b1,  // [D, 1]
                     const int*   __restrict__ flag,
                     float*       __restrict__ out) // [N, D]
{
    __shared__ _Float16 sXhi[ROWS * D_TOW];  // 32 KB x tile, hi plane
    __shared__ _Float16 sXlo[ROWS * D_TOW];  // 32 KB x tile, lo plane
    __shared__ float    sW1[D_HID];
    __shared__ float    sB0[D_HID];
    __shared__ float    sOut[ROWS];          // per-row accumulators

    const int k      = blockIdx.x;           // tower
    const int n_base = blockIdx.y * ROWS;    // sample row base
    const int tid    = threadIdx.x;

    // ---- stage x tile: float4 loads, packed 4xf16 (b64) LDS stores ----
    const float4* xsrc4 = (const float4*)(x + (size_t)n_base * D_TOW);
    for (int i = tid; i < ROWS * D_TOW / 4; i += 256) {
        float4 f = xsrc4[i];
        v4h hi, lo;
        _Float16 h, l;
        f32_split(f.x, h, l); hi[0] = h; lo[0] = l;
        f32_split(f.y, h, l); hi[1] = h; lo[1] = l;
        f32_split(f.z, h, l); hi[2] = h; lo[2] = l;
        f32_split(f.w, h, l); hi[3] = h; lo[3] = l;
        *(v4h*)(sXhi + 4 * i) = hi;
        *(v4h*)(sXlo + 4 * i) = lo;
    }
    if (tid < D_HID) {
        sW1[tid] = W1[k * D_HID + tid];
        sB0[tid] = b0[k * D_HID + tid];
    }
    if (tid < ROWS) sOut[tid] = 0.0f;
    __syncthreads();

    const int wave = tid >> 5;      // 0..7 -> hidden-column tile
    const int lane = tid & 31;
    const int half = lane >> 4;
    const int lm   = lane & 15;
    const int h0   = wave * 16;

    // ---- B operand (W0) in registers for the whole block, hi/lo split.
    // 16-bit B (32x16 KxN) layout: lane = column N=h0+lm, elements 0..7 hold
    // K = 32*kk + 8*half + j, elements 8..15 hold K = 32*kk+16+8*half+(j-8).  8 frags x 8 VGPRs.
    const float* wrow = W0 + (size_t)k * D_HID * D_TOW
                           + (size_t)(h0 + lm) * D_TOW;
    v16h Bhi[KSTEPS], Blo[KSTEPS];
    #pragma unroll
    for (int kk = 0; kk < KSTEPS; ++kk) {
        #pragma unroll
        for (int j = 0; j < 16; ++j) {
            _Float16 hi, lo;
            f32_split(wrow[32 * kk + 8 * half + j + (j >= 8 ? 8 : 0)], hi, lo);
            Bhi[kk][j] = hi;
            Blo[kk][j] = lo;
        }
    }

    const float w1v = sW1[h0 + lm];
    const float b0v = sB0[h0 + lm];

    for (int t = 0; t < TILES; ++t) {
        // 16-bit A (16x32 MxK) layout: lane = row M=lm, elements 0..7 hold
        // K = 32*kk + 8*half + j, elements 8..15 hold K = 32*kk+16+8*half+j.
        const _Float16* xrh = &sXhi[(t * 16 + lm) * D_TOW + 8 * half];
        const _Float16* xrl = &sXlo[(t * 16 + lm) * D_TOW + 8 * half];

        // Two independent WMMA dependency chains; b0 folded into chain 0.
        v8f accH = {b0v, b0v, b0v, b0v, b0v, b0v, b0v, b0v};
        v8f accX = {};
        #pragma unroll
        for (int kk = 0; kk < KSTEPS; ++kk) {
            v8h ah0 = *(const v8h*)(xrh + 32 * kk);        // ds_load_b128
            v8h ah1 = *(const v8h*)(xrh + 32 * kk + 16);
            v8h al0 = *(const v8h*)(xrl + 32 * kk);
            v8h al1 = *(const v8h*)(xrl + 32 * kk + 16);
            v16h Ahi, Alo;
            #pragma unroll
            for (int j = 0; j < 8; ++j) {
                Ahi[j] = ah0[j];  Ahi[j + 8] = ah1[j];
                Alo[j] = al0[j];  Alo[j + 8] = al1[j];
            }
            accH = __builtin_amdgcn_wmma_f32_16x16x32_f16(
                       false, Ahi, false, Bhi[kk], (short)0, accH, false, false);
            accX = __builtin_amdgcn_wmma_f32_16x16x32_f16(
                       false, Ahi, false, Blo[kk], (short)0, accX, false, false);
            accX = __builtin_amdgcn_wmma_f32_16x16x32_f16(
                       false, Alo, false, Bhi[kk], (short)0, accX, false, false);
        }

        // ---- layer 1 epilogue: sigmoid, scale by W1, DPP row reduction.
        // C/D layout: VGPR r -> row (r + 8*half) of tile, col = h0+lm.
        #pragma unroll
        for (int r = 0; r < 8; ++r) {
            float pre = accH[r] + accX[r];
            float v   = fast_sigmoid(pre) * w1v;
            v = row_reduce16(v);                  // VALU-only, no LDS traffic
            if (lm == 15)
                atomicAdd(&sOut[t * 16 + r + 8 * half], v);  // ds_add_f32
        }
    }
    __syncthreads();

    if (tid < ROWS) {
        float o = sOut[tid] + b1[k];
        if (*flag) o = fast_sigmoid(o);
        out[(size_t)(n_base + tid) * D_TOW + k] = o;
    }
}

// ---------------------------------------------------------------------------
extern "C" void kernel_launch(void* const* d_in, const int* in_sizes, int n_in,
                              void* d_out, int out_size, void* d_ws, size_t ws_size,
                              hipStream_t stream) {
    const float* x  = (const float*)d_in[0];
    const float* W0 = (const float*)d_in[1];
    const float* b0 = (const float*)d_in[2];
    const float* W1 = (const float*)d_in[3];
    const float* b1 = (const float*)d_in[4];
    float* out = (float*)d_out;
    int*   flag = (int*)d_ws;

    nm_init_flag<<<1, 1, 0, stream>>>(flag);
    const int nx = in_sizes[0];
    nm_binary_check<<<(nx + 255) / 256, 256, 0, stream>>>(x, nx, flag);

    dim3 grid(D_TOW, N_SAMP / ROWS);   // 128 towers x 64 row-blocks
    nm_fused_kernel<<<grid, 256, 0, stream>>>(x, W0, b0, W1, b1, flag, out);
}
